// MHSA_45054206935085
// MI455X (gfx1250) — matrix operand order repaired and checked by reading, not compiled, not run on hardware
//
#include <hip/hip_runtime.h>
#include <hip/hip_bf16.h>
#include <math.h>

typedef __attribute__((ext_vector_type(16))) _Float16 v16h;
typedef __attribute__((ext_vector_type(8)))  _Float16 v8h;
typedef __attribute__((ext_vector_type(8)))  float    v8f;
typedef __attribute__((ext_vector_type(4)))  float    v4f;
typedef __attribute__((ext_vector_type(4)))  int      v4i;

static constexpr int Bc   = 2;
static constexpr int Cc   = 256;
static constexpr int Lc   = 4096;     // 64*64 spatial tokens
static constexpr int NHc  = 2;
static constexpr int DKc  = 128;
static constexpr int DVc  = 128;

#if defined(__has_builtin)
#if __has_builtin(__builtin_amdgcn_global_load_async_to_lds_b128) && \
    __has_builtin(__builtin_amdgcn_s_wait_asynccnt)
#define USE_ASYNC_LDS 1
#endif
#endif
#ifndef USE_ASYNC_LDS
#define USE_ASYNC_LDS 0
#endif

// builtin signature: (v4i AS1* gaddr, v4i AS3* ldsaddr, imm int offset, imm int cpol)
#define GAS1(p) ((__attribute__((address_space(1))) v4i*)(p))
#define LAS3(p) ((__attribute__((address_space(3))) v4i*)(p))

__device__ inline v16h mk16(v8h lo, v8h hi) {
    v16h r;
#pragma unroll
    for (int j = 0; j < 8; ++j) { r[j] = lo[j]; r[8 + j] = hi[j]; }
    return r;
}

// ---------------------------------------------------------------- f32 -> f16
__global__ __launch_bounds__(256) void cvt_f32_f16(const float* __restrict__ s,
                                                   _Float16* __restrict__ d, int n) {
    int i = blockIdx.x * 256 + threadIdx.x;
    if (i < n) d[i] = (_Float16)s[i];
}

// ------------------------------------------------- x [B,C,L] -> xT [B,L,C] f16
__global__ __launch_bounds__(256) void transpose_x(const float* __restrict__ x,
                                                   _Float16* __restrict__ xT) {
    int i = blockIdx.x * 256 + threadIdx.x;        // total B*L*C, c fastest
    if (i >= Bc * Lc * Cc) return;
    int c = i % Cc;
    int l = (i / Cc) % Lc;
    int b = i / (Cc * Lc);
    xT[i] = (_Float16)x[((size_t)b * Cc + c) * Lc + l];
}

// ----------------------------------------------------------------- QKV GEMM
// grid: (L/128, B*NH, 3).  Out written token-major [l][128].
__global__ __launch_bounds__(256) void qkv_gemm(
    const _Float16* __restrict__ xT,
    const _Float16* __restrict__ Wq16, const _Float16* __restrict__ Wk16,
    const _Float16* __restrict__ Wv16,
    const float* __restrict__ bq, const float* __restrict__ bk,
    const float* __restrict__ bv,
    _Float16* __restrict__ Qo, _Float16* __restrict__ Ko,
    _Float16* __restrict__ Vo) {
    const int nchunk = blockIdx.x;
    const int b  = blockIdx.y / NHc, h = blockIdx.y % NHc;
    const int pj = blockIdx.z;
    const _Float16* W   = (pj == 0 ? Wq16 : pj == 1 ? Wk16 : Wv16) + (size_t)h * DKc * Cc;
    const float*    bia = (pj == 0 ? bq   : pj == 1 ? bk   : bv)   + h * DKc;
    _Float16*       Out = (pj == 0 ? Qo   : pj == 1 ? Ko   : Vo)   +
                          (size_t)(b * NHc + h) * Lc * DKc;
    const int wave = threadIdx.x >> 5, lane = threadIdx.x & 31;
    const int half = lane >> 4, l15 = lane & 15;
    const int mbase = wave * 16;
    const _Float16* xb = xT + (size_t)b * Lc * Cc + (size_t)nchunk * 128 * Cc;

    v16h a[8];
#pragma unroll
    for (int kc = 0; kc < 8; ++kc) {
        const _Float16* ap = W + (size_t)(mbase + l15) * Cc + kc * 32 + half * 8;
        a[kc] = mk16(*(const v8h*)ap, *(const v8h*)(ap + 16));
    }
    v4f b0 = *(const v4f*)(bia + mbase + half * 8);
    v4f b1 = *(const v4f*)(bia + mbase + half * 8 + 4);

#pragma unroll
    for (int nt = 0; nt < 8; ++nt) {
        v8f acc = {};
        const _Float16* bp = xb + (size_t)(nt * 16 + l15) * Cc + half * 8;
#pragma unroll
        for (int kc = 0; kc < 8; ++kc) {
            v16h bf = mk16(*(const v8h*)(bp + kc * 32), *(const v8h*)(bp + kc * 32 + 16));
            acc = __builtin_amdgcn_wmma_f32_16x16x32_f16(false, a[kc], false, bf,
                                                         (short)0, acc, false, false);
        }
        v8h st;
#pragma unroll
        for (int v = 0; v < 8; ++v)
            st[v] = (_Float16)(acc[v] + (v < 4 ? b0[v] : b1[v - 4]));
        int tok = nchunk * 128 + nt * 16 + l15;
        *(v8h*)(Out + (size_t)tok * DKc + mbase + 8 * half) = st;
    }
}

// ----------------------------------------- per-token l2 normalize * sqrt(scale)
__global__ __launch_bounds__(256) void l2norm_scale(_Float16* __restrict__ T,
                                                    const float* __restrict__ scale,
                                                    int total) {
    int idx = blockIdx.x * 256 + threadIdx.x;
    if (idx >= total) return;
    int h = (idx / Lc) % NHc;
    _Float16* p = T + (size_t)idx * DKc;
    v8h buf[16];
    float ss = 0.f;
#pragma unroll
    for (int i = 0; i < 16; ++i) {
        buf[i] = *(const v8h*)(p + i * 8);
#pragma unroll
        for (int j = 0; j < 8; ++j) { float f = (float)buf[i][j]; ss += f * f; }
    }
    float s = sqrtf(scale[h]) / fmaxf(sqrtf(ss), 1e-6f);
#pragma unroll
    for (int i = 0; i < 16; ++i) {
#pragma unroll
        for (int j = 0; j < 8; ++j) buf[i][j] = (_Float16)((float)buf[i][j] * s);
        *(v8h*)(p + i * 8) = buf[i];
    }
}

// ------------------------------------------------------------ flash attention
// grid: (L/128, B*NH). 8 waves/block, each wave owns 16 queries.
// Online softmax; R written token-major [b][l][NH*DV] f16.
__global__ __launch_bounds__(256) void flash_attn(
    const _Float16* __restrict__ Q16, const _Float16* __restrict__ K16,
    const _Float16* __restrict__ V16, _Float16* __restrict__ R16) {
    __shared__ __align__(32) _Float16 ldsK[64 * 128];   // [key][dk]
    __shared__ __align__(32) _Float16 ldsV[128 * 64];   // [dv][key]
    __shared__ __align__(32) _Float16 ldsP[8 * 16 * 64];// per-wave [q][key]

    const int b = blockIdx.y / NHc, h = blockIdx.y % NHc;
    const _Float16* Qg = Q16 + (size_t)(b * NHc + h) * Lc * DKc;
    const _Float16* Kg = K16 + (size_t)(b * NHc + h) * Lc * DKc;
    const _Float16* Vg = V16 + (size_t)(b * NHc + h) * Lc * DVc;
    const int tid = threadIdx.x;
    const int wave = tid >> 5, lane = tid & 31, half = lane >> 4, l15 = lane & 15;
    const int qb = blockIdx.x * 128 + wave * 16;

    v16h qa[4];
#pragma unroll
    for (int kc = 0; kc < 4; ++kc) {
        const _Float16* ap = Qg + (size_t)(qb + l15) * DKc + kc * 32 + half * 8;
        qa[kc] = mk16(*(const v8h*)ap, *(const v8h*)(ap + 16));
    }

    float mrow[8], lrow[8];
    v8f accO[8];
#pragma unroll
    for (int v = 0; v < 8; ++v) { mrow[v] = -3.0e38f; lrow[v] = 0.f; accO[v] = (v8f){}; }
    _Float16* myP = ldsP + wave * 16 * 64;

    for (int kt = 0; kt < Lc / 64; ++kt) {
        __syncthreads();
        {
#if USE_ASYNC_LDS
            // K tile: direct global -> LDS DMA, no VGPR round trip (ASYNCcnt)
            char* srcb = (char*)(Kg + (size_t)kt * 64 * DKc);
            char* dstb = (char*)ldsK;
#pragma unroll
            for (int i = 0; i < 4; ++i)
                __builtin_amdgcn_global_load_async_to_lds_b128(
                    GAS1(srcb + (size_t)(tid + i * 256) * 16),
                    LAS3(dstb + (size_t)(tid + i * 256) * 16), 0, 0);
#else
            const uint4* src = (const uint4*)(Kg + (size_t)kt * 64 * DKc);
            uint4* dst = (uint4*)ldsK;
#pragma unroll
            for (int i = 0; i < 4; ++i) dst[tid + i * 256] = src[tid + i * 256];
#endif
            // stage V tile with transpose: global token-major -> LDS [dv][key]
            int tok = tid >> 2, dvb = (tid & 3) * 32;
            const _Float16* vp = Vg + (size_t)(kt * 64 + tok) * DVc + dvb;
#pragma unroll
            for (int c4 = 0; c4 < 4; ++c4) {
                v8h vv = *(const v8h*)(vp + c4 * 8);
#pragma unroll
                for (int j = 0; j < 8; ++j) ldsV[(dvb + c4 * 8 + j) * 64 + tok] = vv[j];
            }
            // warm L2 for next tile
            if (kt + 1 < Lc / 64) {
                __builtin_prefetch(Kg + (size_t)(kt + 1) * 64 * DKc + tid * 32, 0, 0);
                __builtin_prefetch(Vg + (size_t)(kt + 1) * 64 * DVc + tid * 32, 0, 0);
            }
        }
#if USE_ASYNC_LDS
        __builtin_amdgcn_s_wait_asynccnt(0);
#endif
        __syncthreads();

        // S = Qn * Kn^T : 16 x 64
        v8f sacc[4];
#pragma unroll
        for (int nt = 0; nt < 4; ++nt) {
            sacc[nt] = (v8f){};
            const _Float16* kb = ldsK + (nt * 16 + l15) * 128 + half * 8;
#pragma unroll
            for (int kc = 0; kc < 4; ++kc) {
                v16h bf = mk16(*(const v8h*)(kb + kc * 32), *(const v8h*)(kb + kc * 32 + 16));
                sacc[nt] = __builtin_amdgcn_wmma_f32_16x16x32_f16(
                    false, qa[kc], false, bf, (short)0, sacc[nt], false, false);
            }
        }

        // online softmax (row r = v + 8*half lives across 16 lanes in vgpr v)
        float alpha[8];
#pragma unroll
        for (int v = 0; v < 8; ++v) {
            float mx = fmaxf(fmaxf(sacc[0][v], sacc[1][v]),
                             fmaxf(sacc[2][v], sacc[3][v]));
            mx = fmaxf(mx, __shfl_xor(mx, 1));
            mx = fmaxf(mx, __shfl_xor(mx, 2));
            mx = fmaxf(mx, __shfl_xor(mx, 4));
            mx = fmaxf(mx, __shfl_xor(mx, 8));
            float nm = fmaxf(mrow[v], mx);
            alpha[v] = __expf(mrow[v] - nm);
            mrow[v] = nm;
            float p0 = __expf(sacc[0][v] - nm);
            float p1 = __expf(sacc[1][v] - nm);
            float p2 = __expf(sacc[2][v] - nm);
            float p3 = __expf(sacc[3][v] - nm);
            int prow = (v + 8 * half) * 64 + l15;
            myP[prow +  0] = (_Float16)p0;
            myP[prow + 16] = (_Float16)p1;
            myP[prow + 32] = (_Float16)p2;
            myP[prow + 48] = (_Float16)p3;
            float ssum = p0 + p1 + p2 + p3;
            ssum += __shfl_xor(ssum, 1);
            ssum += __shfl_xor(ssum, 2);
            ssum += __shfl_xor(ssum, 4);
            ssum += __shfl_xor(ssum, 8);
            lrow[v] = lrow[v] * alpha[v] + ssum;
        }
#pragma unroll
        for (int nt = 0; nt < 8; ++nt)
#pragma unroll
            for (int v = 0; v < 8; ++v) accO[nt][v] *= alpha[v];

        asm volatile("" ::: "memory");  // keep P writes before P reads (wave-local LDS in-order)

        // O += P * V
#pragma unroll
        for (int kc2 = 0; kc2 < 2; ++kc2) {
            const _Float16* pp = myP + l15 * 64 + kc2 * 32 + half * 8;
            v16h pa = mk16(*(const v8h*)pp, *(const v8h*)(pp + 16));
#pragma unroll
            for (int nt = 0; nt < 8; ++nt) {
                const _Float16* vp2 = ldsV + (nt * 16 + l15) * 64 + kc2 * 32 + half * 8;
                v16h vb = mk16(*(const v8h*)vp2, *(const v8h*)(vp2 + 16));
                accO[nt] = __builtin_amdgcn_wmma_f32_16x16x32_f16(
                    false, pa, false, vb, (short)0, accO[nt], false, false);
            }
        }
    }

    // epilogue: divide by row sums, write token-major R (channel = h*128 + dv)
    float inv[8];
#pragma unroll
    for (int v = 0; v < 8; ++v) inv[v] = 1.0f / lrow[v];
#pragma unroll
    for (int nt = 0; nt < 8; ++nt)
#pragma unroll
        for (int v = 0; v < 8; ++v) {
            int tok = qb + v + 8 * half;
            int ch  = h * DVc + nt * 16 + l15;
            R16[((size_t)b * Lc + tok) * (NHc * DVc) + ch] =
                (_Float16)(accO[nt][v] * inv[v]);
        }
}

// ---------------------------------------------- merge conv + bias + residual
// grid: (L/128, B, 2).  out[b][o][l] = Wm.R + bm + x
__global__ __launch_bounds__(256) void merge_gemm(
    const _Float16* __restrict__ Wm16, const _Float16* __restrict__ R16,
    const float* __restrict__ bm, const float* __restrict__ x,
    float* __restrict__ out) {
    const int nchunk = blockIdx.x, b = blockIdx.y, mchunk = blockIdx.z;
    const int wave = threadIdx.x >> 5, lane = threadIdx.x & 31;
    const int half = lane >> 4, l15 = lane & 15;
    const int mbase = mchunk * 128 + wave * 16;
    const _Float16* Rb = R16 + (size_t)b * Lc * Cc + (size_t)nchunk * 128 * Cc;

    v16h a[8];
#pragma unroll
    for (int kc = 0; kc < 8; ++kc) {
        const _Float16* ap = Wm16 + (size_t)(mbase + l15) * Cc + kc * 32 + half * 8;
        a[kc] = mk16(*(const v8h*)ap, *(const v8h*)(ap + 16));
    }
    v4f b0 = *(const v4f*)(bm + mbase + half * 8);
    v4f b1 = *(const v4f*)(bm + mbase + half * 8 + 4);

#pragma unroll
    for (int nt = 0; nt < 8; ++nt) {
        v8f acc = {};
        const _Float16* bp = Rb + (size_t)(nt * 16 + l15) * Cc + half * 8;
#pragma unroll
        for (int kc = 0; kc < 8; ++kc) {
            v16h bf = mk16(*(const v8h*)(bp + kc * 32), *(const v8h*)(bp + kc * 32 + 16));
            acc = __builtin_amdgcn_wmma_f32_16x16x32_f16(false, a[kc], false, bf,
                                                         (short)0, acc, false, false);
        }
        int l = nchunk * 128 + nt * 16 + l15;
#pragma unroll
        for (int v = 0; v < 8; ++v) {
            int o = mbase + 8 * half + v;
            size_t gi = ((size_t)b * Cc + o) * Lc + l;
            out[gi] = acc[v] + (v < 4 ? b0[v] : b1[v - 4]) + x[gi];
        }
    }
}

// ---------------------------------------------------------------------------
extern "C" void kernel_launch(void* const* d_in, const int* in_sizes, int n_in,
                              void* d_out, int out_size, void* d_ws, size_t ws_size,
                              hipStream_t stream) {
    const float* x     = (const float*)d_in[0];
    const float* Wq    = (const float*)d_in[1];
    const float* bq    = (const float*)d_in[2];
    const float* Wk    = (const float*)d_in[3];
    const float* bk    = (const float*)d_in[4];
    const float* Wv    = (const float*)d_in[5];
    const float* bv    = (const float*)d_in[6];
    const float* scale = (const float*)d_in[7];
    const float* Wm    = (const float*)d_in[8];
    const float* bm    = (const float*)d_in[9];
    float* out = (float*)d_out;

    char* ws = (char*)d_ws;
    size_t off = 0;
    auto take = [&](size_t elems) {
        _Float16* p = (_Float16*)(ws + off);
        off += ((elems * sizeof(_Float16) + 255) / 256) * 256;
        return p;
    };
    _Float16* Wq16 = take((size_t)NHc * DKc * Cc);
    _Float16* Wk16 = take((size_t)NHc * DKc * Cc);
    _Float16* Wv16 = take((size_t)NHc * DVc * Cc);
    _Float16* Wm16 = take((size_t)Cc * NHc * DVc);
    _Float16* xT16 = take((size_t)Bc * Lc * Cc);
    _Float16* Q16  = take((size_t)Bc * NHc * Lc * DKc);
    _Float16* K16  = take((size_t)Bc * NHc * Lc * DKc);
    _Float16* V16  = take((size_t)Bc * NHc * Lc * DVc);
    _Float16* R16  = take((size_t)Bc * Lc * NHc * DVc);

    int nW = NHc * DKc * Cc;
    cvt_f32_f16<<<(nW + 255) / 256, 256, 0, stream>>>(Wq, Wq16, nW);
    cvt_f32_f16<<<(nW + 255) / 256, 256, 0, stream>>>(Wk, Wk16, nW);
    cvt_f32_f16<<<(nW + 255) / 256, 256, 0, stream>>>(Wv, Wv16, nW);
    int nWm = Cc * NHc * DVc;
    cvt_f32_f16<<<(nWm + 255) / 256, 256, 0, stream>>>(Wm, Wm16, nWm);

    int nX = Bc * Lc * Cc;
    transpose_x<<<(nX + 255) / 256, 256, 0, stream>>>(x, xT16);

    qkv_gemm<<<dim3(Lc / 128, Bc * NHc, 3), 256, 0, stream>>>(
        xT16, Wq16, Wk16, Wv16, bq, bk, bv, Q16, K16, V16);

    int nTok = Bc * NHc * Lc;
    l2norm_scale<<<(nTok + 255) / 256, 256, 0, stream>>>(Q16, scale, nTok);
    l2norm_scale<<<(nTok + 255) / 256, 256, 0, stream>>>(K16, scale, nTok);

    flash_attn<<<dim3(Lc / 128, Bc * NHc), 256, 0, stream>>>(Q16, K16, V16, R16);

    merge_gemm<<<dim3(Lc / 128, Bc, 2), 256, 0, stream>>>(Wm16, R16, bm, x, out);
}
